// SelfAttention_77515569758681
// MI455X (gfx1250) — matrix operand order repaired and checked by reading, not compiled, not run on hardware
//
#include <hip/hip_runtime.h>
#include <hip/hip_bf16.h>

// ---------------------------------------------------------------------------
// Causal self-attention, flash-attention style, bf16 WMMA on gfx1250 (wave32)
// B=128, T=512, E=384, H=64
// Data movement plan: x read once (100MB), q/k/v bf16 intermediates (24MB),
// scores never touch memory (naive would move ~134GB). Async-to-LDS staging
// (ASYNCcnt) with double buffering hides copy latency behind WMMA.
// ---------------------------------------------------------------------------

#define BATCH 128
#define SEQ   512
#define EMB   384
#define HEAD  64

typedef __attribute__((ext_vector_type(16))) __bf16        v16bf;
typedef __attribute__((ext_vector_type(8)))  float         v8f;
typedef __attribute__((ext_vector_type(8)))  unsigned int  v8u;

__device__ __forceinline__ unsigned short f2bf(float f) {
    unsigned u = __builtin_bit_cast(unsigned, f);
    unsigned r = u + 0x7FFFu + ((u >> 16) & 1u);   // round-to-nearest-even
    return (unsigned short)(r >> 16);
}

// Async global->LDS copy of 8 bytes (GLOBAL_LOAD_ASYNC_TO_LDS_B64, ASYNCcnt).
// ldsoff = LDS byte offset (low 32 bits of the generic pointer).
__device__ __forceinline__ void async_copy8(unsigned ldsoff, const void* gsrc) {
    asm volatile("global_load_async_to_lds_b64 %0, %1, off"
                 :: "v"(ldsoff), "v"(gsrc)
                 : "memory");
}
__device__ __forceinline__ void wait_async0() {
    asm volatile("s_wait_asynccnt 0" ::: "memory");
}
__device__ __forceinline__ unsigned lds_off(const void* p) {
    return (unsigned)(size_t)p;            // LDS aperture: addr[31:0] = offset
}

// ===========================================================================
// Kernel 0: one-shot weight conversion fp32 -> bf16, transposed [n=0..191][k]
// so the projection kernel stages tiles as pure async copies (no VALU).
// ===========================================================================
__global__ __launch_bounds__(256) void wconv_kernel(
    const float* __restrict__ Wq, const float* __restrict__ Wk,
    const float* __restrict__ Wv, unsigned short* __restrict__ wt_all)
{
    int idx = blockIdx.x * 256 + threadIdx.x;  // 73728 total
    int m   = idx / 24576;                     // matrix 0..2
    int rem = idx % 24576;
    int k   = rem >> 6;                        // 0..383
    int n   = rem & 63;
    const float* W = (m == 0) ? Wq : ((m == 1) ? Wk : Wv);
    wt_all[(m * 64 + n) * EMB + k] = f2bf(W[k * HEAD + n]);
}

// ===========================================================================
// Kernel 1: fused QKV projection. x[BT,384] fp32 -> q,k (bf16 [BT,64], q
// pre-scaled by E^-0.5) and v stored transposed per batch: vt[b][h][t].
// 2048 blocks x 256 threads; 32 rows x (q|k|v = 192 cols) per block.
// Weight tiles double-buffered via async-to-LDS.
// ===========================================================================
__global__ __launch_bounds__(256) void qkv_proj_kernel(
    const float* __restrict__ x, const unsigned short* __restrict__ wt_all,
    const float* __restrict__ bq, const float* __restrict__ bk,
    const float* __restrict__ bv,
    unsigned short* __restrict__ qb, unsigned short* __restrict__ kb,
    unsigned short* __restrict__ vt)
{
    __shared__ unsigned short xs[32 * 392];     // 32 rows x 384 (+8 pad)
    __shared__ unsigned short wt[2][192 * 36];  // [n][k0..31] (+4 pad), 2 bufs

    const int tid  = threadIdx.x;
    const int lane = tid & 31;
    const int wave = tid >> 5;                  // 0..7
    const int half = lane >> 4;
    const int ln   = lane & 15;
    const int row0 = blockIdx.x * 32;

    // ---- stage x tile (fp32 -> bf16 conversion, once per element) ---------
    {
        const float4* xg = (const float4*)(x + (size_t)row0 * EMB);
        #pragma unroll
        for (int t = 0; t < 12; ++t) {
            int idx = tid + t * 256;            // 3072 float4
            float4 f = xg[idx];
            int e = idx * 4, r = e / EMB, c = e % EMB;
            unsigned int lo = (unsigned)f2bf(f.x) | ((unsigned)f2bf(f.y) << 16);
            unsigned int hi = (unsigned)f2bf(f.z) | ((unsigned)f2bf(f.w) << 16);
            *(uint2*)&xs[r * 392 + c] = make_uint2(lo, hi);
        }
    }

    // ---- async weight tile stager (pure copy, 12.3KB per K-step) ----------
    auto stage_wt = [&](int ks, int buf) {
        #pragma unroll
        for (int t = 0; t < 6; ++t) {
            int idx = tid + t * 256;            // 1536 x 8B
            int n = idx >> 3, c4 = (idx & 7) * 4;
            async_copy8(lds_off(&wt[buf][n * 36 + c4]),
                        wt_all + n * EMB + ks * 32 + c4);
        }
    };
    stage_wt(0, 0);

    const int nb = wave >> 1;                   // 48-col slab of 192
    const int m0 = (wave & 1) * 16;
    v8f acc[3] = {};

    for (int ks = 0; ks < 12; ++ks) {
        wait_async0();                          // tile ks arrived (this wave)
        __syncthreads();                        // all waves arrived & done reading
        if (ks < 11) stage_wt(ks + 1, (ks + 1) & 1);   // overlap next copy
        const unsigned short* wb = wt[ks & 1];

        // A fragment: 16x32 bf16 from xs
        v8u au;
        #pragma unroll
        for (int v = 0; v < 8; ++v) {
            int kk = half * 8 + (v & 3) * 2 + (v >> 2) * 16;
            au[v] = *(const unsigned int*)&xs[(m0 + ln) * 392 + ks * 32 + kk];
        }
        v16bf a = __builtin_bit_cast(v16bf, au);

        #pragma unroll
        for (int t = 0; t < 3; ++t) {
            int n = nb * 48 + t * 16 + ln;
            v8u bu;
            #pragma unroll
            for (int v = 0; v < 8; ++v) {
                int kk = half * 8 + (v & 3) * 2 + (v >> 2) * 16;
                bu[v] = *(const unsigned int*)&wb[n * 36 + kk];
            }
            v16bf bfrag = __builtin_bit_cast(v16bf, bu);
            acc[t] = __builtin_amdgcn_wmma_f32_16x16x32_bf16(
                false, a, false, bfrag, (short)0, acc[t], false, false);
        }
    }

    // ---- bias + store (q scaled by E^-0.5; v transposed) ------------------
    const float scale = 0.05103103630798288f;   // 1/sqrt(384)
    #pragma unroll
    for (int t = 0; t < 3; ++t) {
        int c   = nb * 48 + t * 16 + ln;
        int mtx = c >> 6;                       // 0=q 1=k 2=v (uniform per tile)
        int h   = c & 63;
        const float* bb = (mtx == 0) ? bq : ((mtx == 1) ? bk : bv);
        float bias = bb[h];
        #pragma unroll
        for (int r = 0; r < 8; ++r) {
            int grow = row0 + m0 + half * 8 + r;
            float val = acc[t][r] + bias;
            if (mtx == 0) {
                qb[(size_t)grow * HEAD + h] = f2bf(val * scale);
            } else if (mtx == 1) {
                kb[(size_t)grow * HEAD + h] = f2bf(val);
            } else {
                int bi = grow >> 9, tl = grow & 511;
                vt[((size_t)bi * HEAD + h) * SEQ + tl] = f2bf(val);
            }
        }
    }
}

// ===========================================================================
// Kernel 2: flash attention. grid (T/64, B), 128 threads = 4 waves.
// One 16-row query M-tile per wave; 64-wide K/V tiles double-buffered via
// async-to-LDS; P converted C-layout -> A-layout through per-wave-private LDS.
// ===========================================================================
__global__ __launch_bounds__(128) void flash_attn_kernel(
    const unsigned short* __restrict__ qb,
    const unsigned short* __restrict__ kb,
    const unsigned short* __restrict__ vt,
    float* __restrict__ out)
{
    __shared__ unsigned short q_lds[64 * 72];      // [qrow][h]
    __shared__ unsigned short k_lds[2][64 * 72];   // [krow][h], 2 buffers
    __shared__ unsigned short v_lds[2][64 * 72];   // [h][krow], 2 buffers
    __shared__ unsigned short p_lds[4 * 16 * 72];  // per-wave private P tile

    const int tid  = threadIdx.x;                  // 0..127
    const int lane = tid & 31;
    const int wave = tid >> 5;                     // 0..3
    const int half = lane >> 4;
    const int ln   = lane & 15;
    const int iq   = blockIdx.x;                   // query tile 0..7
    const int b    = blockIdx.y;
    const int qbase = iq * 64;

    // ---- async stagers ----------------------------------------------------
    auto stage_q = [&]() {
        const unsigned short* src = qb + ((size_t)b * SEQ + qbase) * HEAD;
        #pragma unroll
        for (int t = 0; t < 8; ++t) {
            int idx = tid + t * 128;               // 1024 x 8B
            int e = idx * 4, r = e >> 6, c = e & 63;
            async_copy8(lds_off(&q_lds[r * 72 + c]), src + e);
        }
    };
    auto stage_kv = [&](int j, int buf) {
        const int t0 = j * 64;
        const unsigned short* ksrc = kb + ((size_t)b * SEQ + t0) * HEAD;
        #pragma unroll
        for (int t = 0; t < 8; ++t) {
            int idx = tid + t * 128;
            int e = idx * 4, r = e >> 6, c = e & 63;
            async_copy8(lds_off(&k_lds[buf][r * 72 + c]), ksrc + e);
        }
        #pragma unroll
        for (int t = 0; t < 8; ++t) {
            int idx = tid + t * 128;
            int e = idx * 4, h = e >> 6, c = e & 63;
            async_copy8(lds_off(&v_lds[buf][h * 72 + c]),
                        vt + ((size_t)b * HEAD + h) * SEQ + t0 + c);
        }
    };

    stage_q();
    stage_kv(0, 0);
    wait_async0();
    __syncthreads();

    // ---- persistent Q A-fragments (two K=32 steps over H=64) --------------
    v8u qa[2];
    #pragma unroll
    for (int ks = 0; ks < 2; ++ks)
        #pragma unroll
        for (int v = 0; v < 8; ++v) {
            int kk = half * 8 + (v & 3) * 2 + (v >> 2) * 16 + ks * 32;
            qa[ks][v] = *(const unsigned int*)&q_lds[(wave * 16 + ln) * 72 + kk];
        }

    float mst[8], lst[8];
    #pragma unroll
    for (int r = 0; r < 8; ++r) { mst[r] = -__builtin_inff(); lst[r] = 0.0f; }
    v8f oacc[4] = {};

    for (int j = 0; j <= iq; ++j) {
        const int t0  = j * 64;
        const int buf = j & 1;
        if (j < iq) stage_kv(j + 1, (j + 1) & 1);  // overlap next tile copy

        // ---- S = Q K^T (pre-scaled), 4 x 16x16 C tiles --------------------
        v8f s[4] = {};
        #pragma unroll
        for (int ks = 0; ks < 2; ++ks) {
            v16bf a = __builtin_bit_cast(v16bf, qa[ks]);
            #pragma unroll
            for (int st = 0; st < 4; ++st) {
                v8u bu;
                #pragma unroll
                for (int v = 0; v < 8; ++v) {
                    int kk = half * 8 + (v & 3) * 2 + (v >> 2) * 16 + ks * 32;
                    bu[v] = *(const unsigned int*)&k_lds[buf][(st * 16 + ln) * 72 + kk];
                }
                v16bf bfrag = __builtin_bit_cast(v16bf, bu);
                s[st] = __builtin_amdgcn_wmma_f32_16x16x32_bf16(
                    false, a, false, bfrag, (short)0, s[st], false, false);
            }
        }

        // ---- causal mask on diagonal tile ---------------------------------
        if (j == iq) {
            #pragma unroll
            for (int st = 0; st < 4; ++st) {
                int tk = t0 + st * 16 + ln;
                #pragma unroll
                for (int r = 0; r < 8; ++r) {
                    int mq = qbase + wave * 16 + half * 8 + r;
                    if (tk > mq) s[st][r] = -__builtin_inff();
                }
            }
        }

        // ---- online softmax (row r lives at VGPR index r across 16 lanes) -
        float alpha[8];
        #pragma unroll
        for (int r = 0; r < 8; ++r) {
            float v = fmaxf(fmaxf(s[0][r], s[1][r]), fmaxf(s[2][r], s[3][r]));
            v = fmaxf(v, __shfl_xor(v, 1, 32));
            v = fmaxf(v, __shfl_xor(v, 2, 32));
            v = fmaxf(v, __shfl_xor(v, 4, 32));
            v = fmaxf(v, __shfl_xor(v, 8, 32));
            float mnew = fmaxf(mst[r], v);
            alpha[r] = __expf(mst[r] - mnew);
            mst[r] = mnew;
            #pragma unroll
            for (int st = 0; st < 4; ++st) s[st][r] = __expf(s[st][r] - mnew);
            float rs = (s[0][r] + s[1][r]) + (s[2][r] + s[3][r]);
            rs += __shfl_xor(rs, 1, 32);
            rs += __shfl_xor(rs, 2, 32);
            rs += __shfl_xor(rs, 4, 32);
            rs += __shfl_xor(rs, 8, 32);
            lst[r] = lst[r] * alpha[r] + rs;
            #pragma unroll
            for (int ht = 0; ht < 4; ++ht) oacc[ht][r] *= alpha[r];
        }

        // ---- P (C layout) -> per-wave LDS -> A-fragment layout ------------
        unsigned short* pw = &p_lds[wave * 16 * 72];
        #pragma unroll
        for (int st = 0; st < 4; ++st)
            #pragma unroll
            for (int r = 0; r < 8; ++r)
                pw[(half * 8 + r) * 72 + st * 16 + ln] = f2bf(s[st][r]);

        // ---- O += P * V ---------------------------------------------------
        #pragma unroll
        for (int ks = 0; ks < 2; ++ks) {
            v8u pu;
            #pragma unroll
            for (int v = 0; v < 8; ++v) {
                int kk = half * 8 + (v & 3) * 2 + (v >> 2) * 16 + ks * 32;
                pu[v] = *(const unsigned int*)&pw[ln * 72 + kk];
            }
            v16bf pa = __builtin_bit_cast(v16bf, pu);
            #pragma unroll
            for (int ht = 0; ht < 4; ++ht) {
                v8u bu;
                #pragma unroll
                for (int v = 0; v < 8; ++v) {
                    int kk = half * 8 + (v & 3) * 2 + (v >> 2) * 16 + ks * 32;   // t index
                    bu[v] = *(const unsigned int*)&v_lds[buf][(ht * 16 + ln) * 72 + kk];
                }
                v16bf bfrag = __builtin_bit_cast(v16bf, bu);
                oacc[ht] = __builtin_amdgcn_wmma_f32_16x16x32_bf16(
                    false, pa, false, bfrag, (short)0, oacc[ht], false, false);
            }
        }

        if (j < iq) {
            wait_async0();        // next tile present (this wave's asyncs)
            __syncthreads();      // every wave waited -> LDS writes visible
        }
    }

    // ---- normalize and store fp32 output ----------------------------------
    #pragma unroll
    for (int r = 0; r < 8; ++r) {
        float inv = 1.0f / lst[r];
        size_t row = (size_t)b * SEQ + qbase + wave * 16 + half * 8 + r;
        #pragma unroll
        for (int ht = 0; ht < 4; ++ht)
            out[row * HEAD + ht * 16 + ln] = oacc[ht][r] * inv;
    }
}

// ===========================================================================
extern "C" void kernel_launch(void* const* d_in, const int* in_sizes, int n_in,
                              void* d_out, int out_size, void* d_ws, size_t ws_size,
                              hipStream_t stream) {
    const float* x  = (const float*)d_in[0];
    const float* Wq = (const float*)d_in[1];
    const float* bq = (const float*)d_in[2];
    const float* Wk = (const float*)d_in[3];
    const float* bk = (const float*)d_in[4];
    const float* Wv = (const float*)d_in[5];
    const float* bv = (const float*)d_in[6];
    float* out = (float*)d_out;

    // workspace: q, k (row-major bf16), v (transposed bf16), bf16 weights
    const size_t NQKV = (size_t)BATCH * SEQ * HEAD;   // 4M elements each
    unsigned short* qb     = (unsigned short*)d_ws;
    unsigned short* kb     = qb + NQKV;
    unsigned short* vt     = kb + NQKV;
    unsigned short* wt_all = vt + NQKV;               // 3*64*384 = 73728

    wconv_kernel<<<288, 256, 0, stream>>>(Wq, Wk, Wv, wt_all);
    qkv_proj_kernel<<<(BATCH * SEQ) / 32, 256, 0, stream>>>(
        x, wt_all, bq, bk, bv, qb, kb, vt);
    flash_attn_kernel<<<dim3(SEQ / 64, BATCH), 128, 0, stream>>>(
        qb, kb, vt, out);
}
